// EncoderLayer_46316927320320
// MI455X (gfx1250) — compile-verified, matrix operand order checked by reading, not run on hardware
//
#include <hip/hip_runtime.h>
#include <hip/hip_bf16.h>
#include <math.h>

// Problem geometry (fixed by the reference)
#define Bc  4
#define Tc  2048
#define Dc  1024
#define Hc  16
#define Pc  64
#define FFc 4096

typedef __bf16 bf16;
typedef __attribute__((ext_vector_type(16))) __bf16 v16bf;
typedef __attribute__((ext_vector_type(8)))  float  v8f;

// One WMMA 16x32 bf16 operand fragment: 16 bf16 per lane = 2x 16-byte loads.
union FragBF { uint4 u[2]; v16bf v; };

// A/B fragment layout (CDNA5 ISA 7.12.2, 16-bit 16x32):
// lane l: row = l&15 ; half = l>>4
//   elems j=0..7  -> K = k0 + 8*half + j          (contiguous 16B)
//   elems j=8..15 -> K = k0 + 16 + 8*half + (j-8) (contiguous 16B)
__device__ __forceinline__ void load_frag_bf16(FragBF& f, const bf16* __restrict__ base,
                                               int ld, int row, int k0, int half) {
  const bf16* p = base + (size_t)row * (size_t)ld + (size_t)(k0 + 8 * half);
  f.u[0] = *(const uint4*)(p);
  f.u[1] = *(const uint4*)(p + 16);
}

__device__ __forceinline__ void load_frag_f32cvt(FragBF& f, const float* __restrict__ base,
                                                 int ld, int row, int k0, int half) {
  const float* p = base + (size_t)row * (size_t)ld + (size_t)(k0 + 8 * half);
  float4 a0 = *(const float4*)(p);
  float4 a1 = *(const float4*)(p + 4);
  float4 b0 = *(const float4*)(p + 16);
  float4 b1 = *(const float4*)(p + 20);
  f.v[0]  = (bf16)a0.x; f.v[1]  = (bf16)a0.y; f.v[2]  = (bf16)a0.z; f.v[3]  = (bf16)a0.w;
  f.v[4]  = (bf16)a1.x; f.v[5]  = (bf16)a1.y; f.v[6]  = (bf16)a1.z; f.v[7]  = (bf16)a1.w;
  f.v[8]  = (bf16)b0.x; f.v[9]  = (bf16)b0.y; f.v[10] = (bf16)b0.z; f.v[11] = (bf16)b0.w;
  f.v[12] = (bf16)b1.x; f.v[13] = (bf16)b1.y; f.v[14] = (bf16)b1.z; f.v[15] = (bf16)b1.w;
}

// B fragments out of the LDS-staged 64x32 tile (row stride 40 bf16 = 80B,
// so b128 reads by 16 lanes touch distinct bank groups).
#define SROW 40
__device__ __forceinline__ void load_frag_lds(FragBF& f, const bf16* sm, int nlocal, int half) {
  const uint4* p = (const uint4*)(sm + nlocal * SROW + 8 * half);
  f.u[0] = p[0];
  f.u[1] = p[2];   // +32 bytes
}

enum { A_BF16 = 0, A_F32 = 1 };
enum { EPI_F32 = 0, EPI_QK = 1, EPI_VT = 2, EPI_AV = 3, EPI_GELU = 4, EPI_RES = 5 };

// Batched GEMM: C = epi( scale * A[M,K] x BT[N,K]^T ).
// Block tile 256(M) x 64(N): 8 waves x (32x64 wave tile); B tile LDS-staged,
// double-buffered; A fragments stream from global (L2-resident).
template<int ATYPE, int EPI>
__global__ __launch_bounds__(256) void gemm_bt_kernel(
    const void* __restrict__ Aptr, const bf16* __restrict__ BT,
    void* __restrict__ Cptr, int M, int N, int Kd,
    long strideA, long strideBT, float scale,
    const float* __restrict__ bias, const float* __restrict__ resid)
{
  __shared__ __align__(16) bf16 smem[2][64 * SROW];

  const int tid  = threadIdx.x;
  const int lane = tid & 31;
  const int wave = tid >> 5;
  const int half = lane >> 4;
  const int l16  = lane & 15;

  const int tilesM = M >> 8;               // 256-row block tiles
  const int bm = (int)(blockIdx.x % tilesM);
  const int bn = (int)(blockIdx.x / tilesM);
  const int z  = blockIdx.z;

  const bf16*  Ab = (const bf16*)Aptr  + ((ATYPE == A_BF16) ? (long)z * strideA : 0);
  const float* Af = (const float*)Aptr + ((ATYPE == A_F32)  ? (long)z * strideA : 0);
  const bf16*  Bb = BT + (long)z * strideBT;

  const int m0   = bm * 256 + wave * 32;   // this wave's 32-row strip
  const int n0   = bn * 64;
  const int rowA0 = m0 + l16;
  const int rowA1 = m0 + 16 + l16;

  // Staging: thread t copies one 16B chunk: row r=t>>2 (0..63), chunk c=t&3.
  const int srow = tid >> 2;
  const int scol = (tid & 3) * 8;          // bf16 units
  const bf16* gB = Bb + (size_t)(n0 + srow) * (size_t)Kd + scol;
  bf16* sdst0 = &smem[0][srow * SROW + scol];
  bf16* sdst1 = &smem[1][srow * SROW + scol];

  // Prologue: stage k=0 tile.
  uint4 stage = *(const uint4*)(gB);
  *(uint4*)(sdst0) = stage;
  __syncthreads();

  v8f acc[2][4] = {};
  int buf = 0;
  for (int k0 = 0; k0 < Kd; k0 += 32) {
    const bool has_next = (k0 + 32) < Kd;
    if (has_next) stage = *(const uint4*)(gB + k0 + 32);   // global fetch overlaps WMMA

    FragBF a0, a1;
    if constexpr (ATYPE == A_BF16) {
      load_frag_bf16(a0, Ab, Kd, rowA0, k0, half);
      load_frag_bf16(a1, Ab, Kd, rowA1, k0, half);
      if (has_next) {
        __builtin_prefetch((const void*)(Ab + (size_t)rowA0 * Kd + k0 + 32), 0, 0);
        __builtin_prefetch((const void*)(Ab + (size_t)rowA1 * Kd + k0 + 32), 0, 0);
      }
    } else {
      load_frag_f32cvt(a0, Af, Kd, rowA0, k0, half);
      load_frag_f32cvt(a1, Af, Kd, rowA1, k0, half);
    }

    const bf16* sm = smem[buf];
    FragBF b0, b1, b2, b3;
    load_frag_lds(b0, sm, 0 * 16 + l16, half);
    load_frag_lds(b1, sm, 1 * 16 + l16, half);
    load_frag_lds(b2, sm, 2 * 16 + l16, half);
    load_frag_lds(b3, sm, 3 * 16 + l16, half);

    acc[0][0] = __builtin_amdgcn_wmma_f32_16x16x32_bf16(false, a0.v, false, b0.v, (short)0, acc[0][0], false, false);
    acc[0][1] = __builtin_amdgcn_wmma_f32_16x16x32_bf16(false, a0.v, false, b1.v, (short)0, acc[0][1], false, false);
    acc[0][2] = __builtin_amdgcn_wmma_f32_16x16x32_bf16(false, a0.v, false, b2.v, (short)0, acc[0][2], false, false);
    acc[0][3] = __builtin_amdgcn_wmma_f32_16x16x32_bf16(false, a0.v, false, b3.v, (short)0, acc[0][3], false, false);
    acc[1][0] = __builtin_amdgcn_wmma_f32_16x16x32_bf16(false, a1.v, false, b0.v, (short)0, acc[1][0], false, false);
    acc[1][1] = __builtin_amdgcn_wmma_f32_16x16x32_bf16(false, a1.v, false, b1.v, (short)0, acc[1][1], false, false);
    acc[1][2] = __builtin_amdgcn_wmma_f32_16x16x32_bf16(false, a1.v, false, b2.v, (short)0, acc[1][2], false, false);
    acc[1][3] = __builtin_amdgcn_wmma_f32_16x16x32_bf16(false, a1.v, false, b3.v, (short)0, acc[1][3], false, false);

    if (has_next) {
      // buf^1 was last read one iteration ago; that barrier has passed.
      *(uint4*)(buf ? sdst0 : sdst1) = stage;
      __syncthreads();
      buf ^= 1;
    }
  }

  // Epilogue: lane holds C rows m = m0 + sub*16 + 8*half + r, col n = n0 + t*16 + l16.
#pragma unroll
  for (int sub = 0; sub < 2; ++sub) {
#pragma unroll
    for (int t = 0; t < 4; ++t) {
      const int n = n0 + t * 16 + l16;
      float bval = 0.f;
      if constexpr (EPI == EPI_GELU || EPI == EPI_RES) bval = bias[n];
#pragma unroll
      for (int r = 0; r < 8; ++r) {
        const int m = m0 + sub * 16 + half * 8 + r;
        float v = acc[sub][t][r] * scale;
        if constexpr (EPI == EPI_F32) {
          ((float*)Cptr)[(long)z * (long)M * (long)N + (long)m * N + n] = v;
        } else if constexpr (EPI == EPI_QK) {       // -> [b,h,t,p] bf16
          const int b = m >> 11, tt = m & (Tc - 1);
          const int h = n >> 6,  p  = n & (Pc - 1);
          ((bf16*)Cptr)[(((long)(b * Hc + h)) * Tc + tt) * Pc + p] = (bf16)v;
        } else if constexpr (EPI == EPI_VT) {       // -> [b,h,p,t] bf16 (pre-transposed V)
          const int b = m >> 11, tt = m & (Tc - 1);
          const int h = n >> 6,  p  = n & (Pc - 1);
          ((bf16*)Cptr)[(((long)(b * Hc + h)) * Pc + p) * Tc + tt] = (bf16)v;
        } else if constexpr (EPI == EPI_AV) {       // -> new_values [b,t, h*P+p] bf16
          const int b = z / Hc, h = z % Hc;
          ((bf16*)Cptr)[((long)b * Tc + m) * Dc + h * Pc + n] = (bf16)v;
        } else if constexpr (EPI == EPI_GELU) {     // exact gelu, bf16 out
          v += bval;
          v = 0.5f * v * (1.0f + erff(v * 0.70710678118654752f));
          ((bf16*)Cptr)[(long)m * N + n] = (bf16)v;
        } else if constexpr (EPI == EPI_RES) {      // + bias + residual, f32 out
          v += bval + resid[(long)m * N + n];
          ((float*)Cptr)[(long)m * N + n] = v;
        }
      }
    }
  }
}

// LayerNorm over D=1024; optional residual add; writes fp32 and/or bf16.
__global__ __launch_bounds__(256) void ln_kernel(
    const float* __restrict__ in, const float* __restrict__ res,
    const float* __restrict__ g, const float* __restrict__ bsh,
    float* __restrict__ outf, bf16* __restrict__ outb)
{
  __shared__ float red[256];
  const long row = blockIdx.x;
  const float* x = in + row * Dc;
  float lx[4];
  float s = 0.f;
#pragma unroll
  for (int i = 0; i < 4; ++i) { lx[i] = x[threadIdx.x + i * 256]; s += lx[i]; }
  red[threadIdx.x] = s; __syncthreads();
  for (int off = 128; off > 0; off >>= 1) {
    if ((int)threadIdx.x < off) red[threadIdx.x] += red[threadIdx.x + off];
    __syncthreads();
  }
  const float mean = red[0] * (1.0f / Dc);
  __syncthreads();
  float s2 = 0.f;
#pragma unroll
  for (int i = 0; i < 4; ++i) { float d = lx[i] - mean; s2 += d * d; }
  red[threadIdx.x] = s2; __syncthreads();
  for (int off = 128; off > 0; off >>= 1) {
    if ((int)threadIdx.x < off) red[threadIdx.x] += red[threadIdx.x + off];
    __syncthreads();
  }
  const float rstd = rsqrtf(red[0] * (1.0f / Dc) + 1e-5f);
#pragma unroll
  for (int i = 0; i < 4; ++i) {
    const int idx = threadIdx.x + i * 256;
    float v = (lx[i] - mean) * rstd * g[idx] + bsh[idx];
    if (res)  v += res[row * Dc + idx];
    if (outf) outf[row * Dc + idx] = v;
    if (outb) outb[row * Dc + idx] = (bf16)v;
  }
}

// In-place softmax over rows of length T=2048 (one block per row).
__global__ __launch_bounds__(256) void softmax_kernel(float* __restrict__ attn)
{
  __shared__ float red[256];
  const long row = blockIdx.x;
  float* p = attn + row * (long)Tc;
  float lv[8];
  float mx = -3.0e38f;
#pragma unroll
  for (int i = 0; i < 8; ++i) { lv[i] = p[threadIdx.x + i * 256]; mx = fmaxf(mx, lv[i]); }
  red[threadIdx.x] = mx; __syncthreads();
  for (int off = 128; off > 0; off >>= 1) {
    if ((int)threadIdx.x < off) red[threadIdx.x] = fmaxf(red[threadIdx.x], red[threadIdx.x + off]);
    __syncthreads();
  }
  mx = red[0]; __syncthreads();
  float s = 0.f;
#pragma unroll
  for (int i = 0; i < 8; ++i) { lv[i] = __expf(lv[i] - mx); s += lv[i]; }
  red[threadIdx.x] = s; __syncthreads();
  for (int off = 128; off > 0; off >>= 1) {
    if ((int)threadIdx.x < off) red[threadIdx.x] += red[threadIdx.x + off];
    __syncthreads();
  }
  const float inv = 1.0f / red[0];
#pragma unroll
  for (int i = 0; i < 8; ++i) p[threadIdx.x + i * 256] = lv[i] * inv;
}

// WQ/WK/WV (H,D,P) -> BT[n = h*P+p][k = d] bf16
__global__ __launch_bounds__(256) void wqkv_bt_kernel(const float* __restrict__ src,
                                                      bf16* __restrict__ dst)
{
  const long i = (long)blockIdx.x * 256 + threadIdx.x;
  if (i >= (long)Dc * Dc) return;
  const int n = (int)(i / Dc), d = (int)(i % Dc);
  const int h = n >> 6, p = n & (Pc - 1);
  dst[i] = (bf16)src[((long)h * Dc + d) * Pc + p];
}

// Generic [K,N] f32 -> [N,K] bf16 transpose (for WO, fW1, fW2).
__global__ __launch_bounds__(256) void trans_bt_kernel(const float* __restrict__ src,
                                                       bf16* __restrict__ dst, int K, int N)
{
  const long i = (long)blockIdx.x * 256 + threadIdx.x;
  if (i >= (long)K * N) return;
  const int n = (int)(i / K), k = (int)(i % K);
  dst[i] = (bf16)src[(long)k * N + n];
}

static inline dim3 gemm_grid(long M, long N, int z) {
  long blocks = (M / 256) * (N / 64);
  return dim3((unsigned)blocks, 1, (unsigned)z);
}

extern "C" void kernel_launch(void* const* d_in, const int* in_sizes, int n_in,
                              void* d_out, int out_size, void* d_ws, size_t ws_size,
                              hipStream_t stream)
{
  const float* X   = (const float*)d_in[0];
  const float* WQ  = (const float*)d_in[1];
  const float* WK  = (const float*)d_in[2];
  const float* WV  = (const float*)d_in[3];
  const float* WO  = (const float*)d_in[4];
  const float* ag  = (const float*)d_in[5];
  const float* ab  = (const float*)d_in[6];
  const float* fg  = (const float*)d_in[7];
  const float* fb  = (const float*)d_in[8];
  const float* fW1 = (const float*)d_in[9];
  const float* fb1 = (const float*)d_in[10];
  const float* fW2 = (const float*)d_in[11];
  const float* fb2 = (const float*)d_in[12];

  float* out1 = (float*)d_out;                         // [B,T,D] fp32
  float* attn = out1 + (long)Bc * Tc * Dc;             // [B,H,T,T] fp32

  const long BT = (long)Bc * Tc;
  char* w = (char*)d_ws;
  auto carve = [&](size_t bytes) -> void* {
    void* p = (void*)w; w += (bytes + 255) & ~(size_t)255; return p;
  };
  bf16*  nXb = (bf16*)carve(BT * Dc * 2);              // LN(X) bf16
  bf16*  Qb  = (bf16*)carve(BT * Dc * 2);              // [b,h,t,p]
  bf16*  Kb  = (bf16*)carve(BT * Dc * 2);              // [b,h,t,p]
  bf16*  VTb = (bf16*)carve(BT * Dc * 2);              // [b,h,p,t]
  bf16*  NVb = (bf16*)carve(BT * Dc * 2);              // new_values [b,t,h*P+p]
  float* SS  = (float*)carve(BT * Dc * 4);             // seq_summary fp32
  float* PF  = (float*)carve(BT * Dc * 4);             // pre_ff fp32
  bf16*  PFb = (bf16*)carve(BT * Dc * 2);              // pre_ff bf16
  bf16*  Hb  = (bf16*)carve(BT * FFc * 2);             // hidden bf16
  bf16*  WQt = (bf16*)carve((long)Dc * Dc * 2);
  bf16*  WKt = (bf16*)carve((long)Dc * Dc * 2);
  bf16*  WVt = (bf16*)carve((long)Dc * Dc * 2);
  bf16*  WOt = (bf16*)carve((long)Dc * Dc * 2);
  bf16*  W1t = (bf16*)carve((long)Dc * FFc * 2);
  bf16*  W2t = (bf16*)carve((long)Dc * FFc * 2);

  // --- weight convert/transpose (one-time cost, negligible vs 275 GFLOP) ---
  {
    const int blk1 = (int)(((long)Dc * Dc + 255) / 256);
    wqkv_bt_kernel<<<blk1, 256, 0, stream>>>(WQ, WQt);
    wqkv_bt_kernel<<<blk1, 256, 0, stream>>>(WK, WKt);
    wqkv_bt_kernel<<<blk1, 256, 0, stream>>>(WV, WVt);
    trans_bt_kernel<<<blk1, 256, 0, stream>>>(WO, WOt, Dc, Dc);
    const int blk2 = (int)(((long)Dc * FFc + 255) / 256);
    trans_bt_kernel<<<blk2, 256, 0, stream>>>(fW1, W1t, Dc, FFc);
    trans_bt_kernel<<<blk2, 256, 0, stream>>>(fW2, W2t, FFc, Dc);
  }

  // --- nX = LN(X) ---
  ln_kernel<<<(int)BT, 256, 0, stream>>>(X, nullptr, ag, ab, nullptr, nXb);

  // --- QKV projections (Q/K head-major, V pre-transposed) ---
  gemm_bt_kernel<A_BF16, EPI_QK><<<gemm_grid(BT, Dc, 1), 256, 0, stream>>>(
      nXb, WQt, Qb, (int)BT, Dc, Dc, 0, 0, 1.f, nullptr, nullptr);
  gemm_bt_kernel<A_BF16, EPI_QK><<<gemm_grid(BT, Dc, 1), 256, 0, stream>>>(
      nXb, WKt, Kb, (int)BT, Dc, Dc, 0, 0, 1.f, nullptr, nullptr);
  gemm_bt_kernel<A_BF16, EPI_VT><<<gemm_grid(BT, Dc, 1), 256, 0, stream>>>(
      nXb, WVt, VTb, (int)BT, Dc, Dc, 0, 0, 1.f, nullptr, nullptr);

  // --- scores: S[q,k] = K[q]·Q[k] / sqrt(P)  (reference swaps q/k), fp32 -> d_out ---
  gemm_bt_kernel<A_BF16, EPI_F32><<<gemm_grid(Tc, Tc, Bc * Hc), 256, 0, stream>>>(
      Kb, Qb, attn, Tc, Tc, Pc, (long)Tc * Pc, (long)Tc * Pc, 0.125f, nullptr, nullptr);

  // --- softmax in place over last axis ---
  softmax_kernel<<<(int)(Bc * Hc * (long)Tc), 256, 0, stream>>>(attn);

  // --- new_values = attn @ V  (A read fp32 from d_out, converted to bf16 in-register) ---
  gemm_bt_kernel<A_F32, EPI_AV><<<gemm_grid(Tc, Pc, Bc * Hc), 256, 0, stream>>>(
      attn, VTb, NVb, Tc, Pc, Tc, (long)Tc * Tc, (long)Pc * Tc, 1.f, nullptr, nullptr);

  // --- seq_summary = new_values @ WO ---
  gemm_bt_kernel<A_BF16, EPI_F32><<<gemm_grid(BT, Dc, 1), 256, 0, stream>>>(
      NVb, WOt, SS, (int)BT, Dc, Dc, 0, 0, 1.f, nullptr, nullptr);

  // --- pre_ff = X + LN(seq_summary) ---
  ln_kernel<<<(int)BT, 256, 0, stream>>>(SS, X, fg, fb, PF, PFb);

  // --- hidden = gelu(pre_ff @ fW1 + fb1) ---
  gemm_bt_kernel<A_BF16, EPI_GELU><<<gemm_grid(BT, FFc, 1), 256, 0, stream>>>(
      PFb, W1t, Hb, (int)BT, FFc, Dc, 0, 0, 1.f, fb1, nullptr);

  // --- out1 = hidden @ fW2 + fb2 + pre_ff ---
  gemm_bt_kernel<A_BF16, EPI_RES><<<gemm_grid(BT, Dc, 1), 256, 0, stream>>>(
      Hb, W2t, out1, (int)BT, Dc, FFc, 0, 0, 1.f, fb2, PF);
}
